// GNNModel_59880434040858
// MI455X (gfx1250) — matrix operand order repaired and checked by reading, not compiled, not run on hardware
//
#include <hip/hip_runtime.h>

#define F_DIM 128
#define H_DIM 16
#define C_DIM 40

typedef __attribute__((ext_vector_type(16))) __bf16   v16bf;
typedef __attribute__((ext_vector_type(8)))  float    v8f;
typedef __attribute__((ext_vector_type(8)))  unsigned v8u;

union BF16Frag { v8u u; v16bf b; };

__device__ __forceinline__ unsigned f32_to_bf16_rne(float f) {
    unsigned x = __float_as_uint(f);
    return (x + 0x7FFFu + ((x >> 16) & 1u)) >> 16;
}
__device__ __forceinline__ unsigned pk_bf16(float lo, float hi) {
    return f32_to_bf16_rne(lo) | (f32_to_bf16_rne(hi) << 16);
}

// ---------------- degree / normalization ----------------
__global__ void k_init_deg(unsigned* deg, int n) {
    int i = blockIdx.x * blockDim.x + threadIdx.x;
    if (i < n) deg[i] = 1u;                        // self loop
}
__global__ void k_deg_edges(const long long* __restrict__ dst, unsigned* deg, int e) {
    int i = blockIdx.x * blockDim.x + threadIdx.x;
    if (i < e) atomicAdd(&deg[(int)dst[i]], 1u);
}
__global__ void k_dinv(unsigned* deg, int n) {     // in-place: u32 count -> f32 rsqrt bits
    int i = blockIdx.x * blockDim.x + threadIdx.x;
    if (i < n) deg[i] = __float_as_uint(rsqrtf((float)deg[i]));
}

// ---------------- bf16 packing of x (float4 -> uint2) ----------------
__global__ void k_pack_x(const float* __restrict__ x, unsigned* __restrict__ xpk, int cnt) {
    int i = blockIdx.x * blockDim.x + threadIdx.x;   // cnt = N * F/4
    if (i < cnt) {
        float4 v = ((const float4*)x)[i];
        uint2  p = make_uint2(pk_bf16(v.x, v.y), pk_bf16(v.z, v.w));
        ((uint2*)xpk)[i] = p;
    }
}

// ---------------- prepack W1 into B-fragment layout [kc][lane][v] ----------------
__global__ void k_pack_w1(const float* __restrict__ W1, unsigned* __restrict__ w1pk) {
    int i = blockIdx.x * blockDim.x + threadIdx.x;   // 4*32*8 = 1024
    if (i < 1024) {
        int v = i & 7, lane = (i >> 3) & 31, kc = i >> 8;
        int g = lane >> 4, ncol = lane & 15;
        int kb = kc * 32 + (v < 4 ? g * 8 + 2 * v : 16 + g * 8 + 2 * (v - 4));
        w1pk[i] = pk_bf16(W1[kb * H_DIM + ncol], W1[(kb + 1) * H_DIM + ncol]);
    }
}
// ---------------- prepack W2 (K padded to 32, cols padded to 48) [t][lane][v] ----
__global__ void k_pack_w2(const float* __restrict__ W2, unsigned* __restrict__ w2pk) {
    int i = blockIdx.x * blockDim.x + threadIdx.x;   // 3*32*8 = 768
    if (i < 768) {
        int v = i & 7, lane = (i >> 3) & 31, t = i >> 8;
        int g = lane >> 4, ncol = lane & 15;
        int kb = (v < 4 ? g * 8 + 2 * v : 16 + g * 8 + 2 * (v - 4));
        int n  = t * 16 + ncol;
        unsigned val = 0u;
        if (v < 4 && n < C_DIM)                      // K>=16 rows and cols>=40 are zero
            val = pk_bf16(W2[kb * C_DIM + n], W2[(kb + 1) * C_DIM + n]);
        w2pk[i] = val;
    }
}

// ---------------- GEMM1: h1[N,16] = x[N,128] @ W1[128,16] (bf16 WMMA, f32 acc) ----
__global__ void k_gemm1(const unsigned* __restrict__ xpk, const unsigned* __restrict__ w1pk,
                        float* __restrict__ h1, int ntiles) {
    int wave = threadIdx.x >> 5;
    int lane = threadIdx.x & 31;
    int tile = blockIdx.x * 4 + wave;
    if (tile >= ntiles) return;                      // wave-uniform exit
    int g    = lane >> 4;
    int m    = lane & 15;
    const uint4* arow = (const uint4*)(xpk + (size_t)(tile * 16 + m) * (F_DIM / 2));

    v8f c = {};
#pragma unroll
    for (int kc = 0; kc < 4; ++kc) {
        uint4 alo = arow[kc * 4 + g];                // dwords k/2 = kc*16 + 4g + 0..3
        uint4 ahi = arow[kc * 4 + 2 + g];            // dwords k/2 = kc*16 + 8 + 4g + 0..3
        const uint4* bp = (const uint4*)(w1pk + (size_t)(kc * 32 + lane) * 8);
        uint4 blo = bp[0];
        uint4 bhi = bp[1];
        BF16Frag a, b;
        a.u = (v8u){alo.x, alo.y, alo.z, alo.w, ahi.x, ahi.y, ahi.z, ahi.w};
        b.u = (v8u){blo.x, blo.y, blo.z, blo.w, bhi.x, bhi.y, bhi.z, bhi.w};
        c = __builtin_amdgcn_wmma_f32_16x16x32_bf16(false, a.b, false, b.b,
                                                    (short)0, c, false, false);
    }
#pragma unroll
    for (int v = 0; v < 8; ++v)                      // full tiles: no guards
        h1[(size_t)(tile * 16 + v + 8 * g) * H_DIM + (lane & 15)] = c[v];
}

// scalar tail for rows not covered by full tiles (empty when n % 16 == 0)
__global__ void k_gemm1_tail(const float* __restrict__ x, const float* __restrict__ W1,
                             float* __restrict__ h1, int r0, int n) {
    int i = blockIdx.x * blockDim.x + threadIdx.x;   // (n-r0)*16
    int row = r0 + (i >> 4), col = i & 15;
    if (row < n) {
        float s = 0.f;
        for (int k = 0; k < F_DIM; ++k) s += x[row * F_DIM + k] * W1[k * H_DIM + col];
        h1[row * H_DIM + col] = s;
    }
}

// ---------------- layer1 aggregate ----------------
__global__ void k_init_agg1(const float* __restrict__ h1, const float* __restrict__ dinv,
                            const float* __restrict__ b1, float* __restrict__ agg1, int cnt) {
    int i = blockIdx.x * blockDim.x + threadIdx.x;   // cnt = N*4
    if (i < cnt) {
        int node = i >> 2, q = i & 3;
        float di = dinv[node], d2 = di * di;
        float4 h = ((const float4*)h1)[i];
        float4 bb = ((const float4*)b1)[q];
        float4 r = make_float4(bb.x + h.x * d2, bb.y + h.y * d2,
                               bb.z + h.z * d2, bb.w + h.w * d2);
        ((float4*)agg1)[i] = r;
    }
}
__global__ void k_scatter1(const long long* __restrict__ src, const long long* __restrict__ dst,
                           const float* __restrict__ dinv, const float* __restrict__ h1,
                           float* __restrict__ agg1, int cnt) {
    int i = blockIdx.x * blockDim.x + threadIdx.x;   // cnt = E*4
    if (i < cnt) {
        int e = i >> 2, q = (i & 3) * 4;
        int s = (int)src[e], d = (int)dst[e];
        float w = dinv[s] * dinv[d];
        float4 v = *(const float4*)(h1 + (size_t)s * H_DIM + q);
        float* o = agg1 + (size_t)d * H_DIM + q;
        unsafeAtomicAdd(o + 0, v.x * w);
        unsafeAtomicAdd(o + 1, v.y * w);
        unsafeAtomicAdd(o + 2, v.z * w);
        unsafeAtomicAdd(o + 3, v.w * w);
    }
}
__global__ void k_relu_pack(const float* __restrict__ agg1, unsigned* __restrict__ zpk, int cnt) {
    int i = blockIdx.x * blockDim.x + threadIdx.x;   // cnt = N*4
    if (i < cnt) {
        float4 v = ((const float4*)agg1)[i];
        uint2  p = make_uint2(pk_bf16(fmaxf(v.x, 0.f), fmaxf(v.y, 0.f)),
                              pk_bf16(fmaxf(v.z, 0.f), fmaxf(v.w, 0.f)));
        ((uint2*)zpk)[i] = p;
    }
}

// ---------------- GEMM2: h2[N,40] = z[N,16] @ W2[16,40] (K padded to 32) -------
__global__ void k_gemm2(const unsigned* __restrict__ zpk, const unsigned* __restrict__ w2pk,
                        float* __restrict__ h2, int ntiles) {
    int wave = threadIdx.x >> 5;
    int lane = threadIdx.x & 31;
    int tile = blockIdx.x * 4 + wave;
    if (tile >= ntiles) return;
    int g    = lane >> 4;
    int m    = lane & 15;
    int ncol = lane & 15;

    uint4 a0 = *(const uint4*)(zpk + (size_t)(tile * 16 + m) * (H_DIM / 2) + 4 * g);
    BF16Frag a;
    a.u = (v8u){a0.x, a0.y, a0.z, a0.w, 0u, 0u, 0u, 0u};

#pragma unroll
    for (int t = 0; t < 3; ++t) {
        const uint4* bp = (const uint4*)(w2pk + (size_t)(t * 32 + lane) * 8);
        uint4 blo = bp[0];
        uint4 bhi = bp[1];
        BF16Frag b;
        b.u = (v8u){blo.x, blo.y, blo.z, blo.w, bhi.x, bhi.y, bhi.z, bhi.w};
        v8f c = {};
        c = __builtin_amdgcn_wmma_f32_16x16x32_bf16(false, a.b, false, b.b,
                                                    (short)0, c, false, false);
        int ncg = t * 16 + ncol;
        bool ok = (t < 2) | (ncol < (C_DIM - 32));   // folds to unguarded for t=0,1
        if (ok) {
#pragma unroll
            for (int v = 0; v < 8; ++v)
                h2[(size_t)(tile * 16 + v + 8 * g) * C_DIM + ncg] = c[v];
        }
    }
}
__global__ void k_gemm2_tail(const float* __restrict__ agg1, const float* __restrict__ W2,
                             float* __restrict__ h2, int r0, int n) {
    int i = blockIdx.x * blockDim.x + threadIdx.x;   // (n-r0)*40
    int row = r0 + i / C_DIM, col = i % C_DIM;
    if (row < n) {
        float s = 0.f;
        for (int k = 0; k < H_DIM; ++k)
            s += fmaxf(agg1[row * H_DIM + k], 0.f) * W2[k * C_DIM + col];
        h2[row * C_DIM + col] = s;
    }
}

// ---------------- layer2 aggregate ----------------
__global__ void k_init_out(const float* __restrict__ h2, const float* __restrict__ dinv,
                           const float* __restrict__ b2, float* __restrict__ out, int cnt) {
    int i = blockIdx.x * blockDim.x + threadIdx.x;   // cnt = N*10
    if (i < cnt) {
        int node = i / 10, q = i % 10;
        float di = dinv[node], d2 = di * di;
        float4 h = ((const float4*)h2)[i];
        float4 bb = ((const float4*)b2)[q];
        float4 r = make_float4(bb.x + h.x * d2, bb.y + h.y * d2,
                               bb.z + h.z * d2, bb.w + h.w * d2);
        ((float4*)out)[i] = r;
    }
}
__global__ void k_scatter2(const long long* __restrict__ src, const long long* __restrict__ dst,
                           const float* __restrict__ dinv, const float* __restrict__ h2,
                           float* __restrict__ out, int cnt) {
    int i = blockIdx.x * blockDim.x + threadIdx.x;   // cnt = E*10
    if (i < cnt) {
        int e = i / 10, q = (i % 10) * 4;
        int s = (int)src[e], d = (int)dst[e];
        float w = dinv[s] * dinv[d];
        float4 v = *(const float4*)(h2 + (size_t)s * C_DIM + q);
        float* o = out + (size_t)d * C_DIM + q;
        unsafeAtomicAdd(o + 0, v.x * w);
        unsafeAtomicAdd(o + 1, v.y * w);
        unsafeAtomicAdd(o + 2, v.z * w);
        unsafeAtomicAdd(o + 3, v.w * w);
    }
}

static inline int div_up(long long a, int b) { return (int)((a + b - 1) / b); }

extern "C" void kernel_launch(void* const* d_in, const int* in_sizes, int n_in,
                              void* d_out, int out_size, void* d_ws, size_t ws_size,
                              hipStream_t stream) {
    const float*     x   = (const float*)d_in[0];
    const long long* ei  = (const long long*)d_in[1];
    const float*     W1  = (const float*)d_in[2];
    const float*     b1  = (const float*)d_in[3];
    const float*     W2  = (const float*)d_in[4];
    const float*     b2  = (const float*)d_in[5];
    float*           out = (float*)d_out;

    const int       n = in_sizes[0] / F_DIM;          // 100000
    const long long e = in_sizes[1] / 2;              // 3200000
    const long long* src = ei;
    const long long* dst = ei + e;

    // workspace layout (256B aligned slabs)
    char*  ws  = (char*)d_ws;
    size_t off = 0;
    auto alloc = [&](size_t bytes) {
        char* p = ws + off;
        off += (bytes + 255) & ~(size_t)255;
        return p;
    };
    unsigned* deg  = (unsigned*)alloc((size_t)n * 4);                 // -> dinv (f32 bits)
    unsigned* xpk  = (unsigned*)alloc((size_t)n * (F_DIM / 2) * 4);   // bf16-packed x
    float*    h1   = (float*)   alloc((size_t)n * H_DIM * 4);
    float*    agg1 = (float*)   alloc((size_t)n * H_DIM * 4);
    unsigned* zpk  = (unsigned*)alloc((size_t)n * (H_DIM / 2) * 4);
    unsigned* w1pk = (unsigned*)alloc(1024 * 4);
    unsigned* w2pk = (unsigned*)alloc(768 * 4);
    float*    h2   = (float*)xpk;           // alias: xpk dead after gemm1; 16MB <= 25.6MB
    const float* dinv = (const float*)deg;

    const int B = 256;
    const int ntiles = n / 16;              // full tiles only
    const int r0     = ntiles * 16;
    const int tail   = n - r0;              // 0 for N=100000

    k_init_deg <<<div_up(n, B), B, 0, stream>>>(deg, n);
    k_deg_edges<<<div_up(e, B), B, 0, stream>>>(dst, deg, (int)e);
    k_dinv     <<<div_up(n, B), B, 0, stream>>>(deg, n);

    k_pack_x <<<div_up((long long)n * (F_DIM / 4), B), B, 0, stream>>>(x, xpk, n * (F_DIM / 4));
    k_pack_w1<<<4, B, 0, stream>>>(W1, w1pk);
    k_pack_w2<<<3, B, 0, stream>>>(W2, w2pk);

    if (ntiles > 0) k_gemm1<<<div_up(ntiles, 4), 128, 0, stream>>>(xpk, w1pk, h1, ntiles);
    if (tail > 0)   k_gemm1_tail<<<div_up((long long)tail * 16, B), B, 0, stream>>>(x, W1, h1, r0, n);

    k_init_agg1<<<div_up((long long)n * 4, B), B, 0, stream>>>(h1, dinv, b1, agg1, n * 4);
    k_scatter1 <<<div_up(e * 4, B), B, 0, stream>>>(src, dst, dinv, h1, agg1, (int)(e * 4));
    k_relu_pack<<<div_up((long long)n * 4, B), B, 0, stream>>>(agg1, zpk, n * 4);

    if (ntiles > 0) k_gemm2<<<div_up(ntiles, 4), 128, 0, stream>>>(zpk, w2pk, h2, ntiles);
    if (tail > 0)   k_gemm2_tail<<<div_up((long long)tail * C_DIM, B), B, 0, stream>>>(agg1, W2, h2, r0, n);

    k_init_out<<<div_up((long long)n * 10, B), B, 0, stream>>>(h2, dinv, b2, out, n * 10);
    k_scatter2<<<div_up(e * 10, B), B, 0, stream>>>(src, dst, dinv, h2, out, (int)(e * 10));
}
